// DeepEVDModel_7464653160723
// MI455X (gfx1250) — matrix operand order repaired and checked by reading, not compiled, not run on hardware
//
#include <hip/hip_runtime.h>
#include <hip/hip_bf16.h>
#include <math.h>

// ---------------------------------------------------------------------------
// DeepEVD on MI455X (gfx1250): all contractions via v_wmma_f32_16x16x32_bf16.
// f32 params/activations cast to bf16 operands (f32 accumulate). Attention is
// flash-style with K tiles staged by the Tensor Data Mover (TDM) when the
// builtin is available; the sequential LSTM runs as ONE persistent 32-wave
// workgroup with state in the 320KB WGP LDS. Wave-uniform tile coordinates are
// forced through readfirstlane so WMMA never sits under exec-masked control
// flow (ISA 7.12: EXEC must be all-1s).
// ---------------------------------------------------------------------------

#define GF_BIAS     1
#define GF_RELU     2
#define GF_SIGMOID  4
#define GF_SOFTPLUS 8
#define GF_OUTF32   16
#define GF_OUTBF16  32

typedef __attribute__((ext_vector_type(16))) __bf16          v16bf;
typedef __attribute__((ext_vector_type(8)))  float           v8f;
typedef __attribute__((ext_vector_type(8)))  unsigned short  us8;
typedef __attribute__((ext_vector_type(4)))  unsigned int    u32x4;
typedef __attribute__((ext_vector_type(8)))  int             i32x8;
typedef __attribute__((ext_vector_type(4)))  int             i32x4;

#if defined(__HIP_DEVICE_COMPILE__)
#if __has_builtin(__builtin_amdgcn_tensor_load_to_lds)
#define HAVE_TDM 1
#endif
#endif

static __device__ __forceinline__ int rfl(int x) {
  return __builtin_amdgcn_readfirstlane(x);   // wave-uniform -> SGPR
}

static __device__ __forceinline__ unsigned short f2bf(float f) {
  unsigned int u = __float_as_uint(f);
  u += 0x7fffu + ((u >> 16) & 1u);            // round-to-nearest-even
  return (unsigned short)(u >> 16);
}

static __device__ __forceinline__ float sigm(float x) { return 1.f / (1.f + expf(-x)); }

static __device__ __forceinline__ v8f wmma_bf(v16bf a, v16bf b, v8f c) {
  // D(f32 16x16) = A(16x32 bf16) * B(32x16 bf16) + C
  return __builtin_amdgcn_wmma_f32_16x16x32_bf16(false, a, false, b, (short)0, c,
                                                 false, false);
}

#ifdef HAVE_TDM
// Issue a TDM 2D tile load (global -> LDS), D# per ISA ch.8: group0 holds
// count/lds_addr/global_addr/type, group1 holds dims/tile/strides.
static __device__ __forceinline__ void tdm_load_2d(unsigned lds_off,
                                                   const void* gptr,
                                                   unsigned tile_w, unsigned tile_h,
                                                   unsigned tensor_w, unsigned tensor_h,
                                                   unsigned long long row_stride) {
  unsigned long long ga = (unsigned long long)gptr;
  u32x4 g0;
  g0[0] = 1u;                                              // count=1, user, load
  g0[1] = lds_off;                                         // lds_addr (bytes)
  g0[2] = (unsigned)ga;                                    // global_addr[31:0]
  g0[3] = (unsigned)((ga >> 32) & 0x01FFFFFFu) | (2u << 30); // addr[56:32]|type=2
  i32x8 g1;
  g1[0] = (int)(1u << 16);                                 // data_size = 2 bytes
  g1[1] = (int)((tensor_w & 0xFFFFu) << 16);               // tensor_dim0 lo
  g1[2] = (int)(((tensor_w >> 16) & 0xFFFFu) | ((tensor_h & 0xFFFFu) << 16));
  g1[3] = (int)(((tensor_h >> 16) & 0xFFFFu) | ((tile_w & 0xFFFFu) << 16));
  g1[4] = (int)(tile_h & 0xFFFFu);                         // tile_dim1; dim2=0
  g1[5] = (int)(unsigned)(row_stride & 0xFFFFFFFFull);     // dim0_stride lo32
  g1[6] = (int)(unsigned)((row_stride >> 32) & 0xFFFFull); // dim0_stride hi16
  g1[7] = 0;
  i32x4 gz = {0, 0, 0, 0};
#if defined(__clang_major__) && (__clang_major__ >= 23)
  i32x8 gz8 = {0, 0, 0, 0, 0, 0, 0, 0};
  __builtin_amdgcn_tensor_load_to_lds(g0, g1, gz, gz, gz8, 0);
#else
  __builtin_amdgcn_tensor_load_to_lds(g0, g1, gz, gz, 0);
#endif
}
#endif

static __device__ __forceinline__ void tdm_wait() {
#ifdef HAVE_TDM
  __builtin_amdgcn_s_wait_tensorcnt(0);
#endif
}

// A fragment (16x32): lane L<16 holds row m0+L, K=kb+[0..15];
// lane L>=16 holds K=kb+[16..31] (contiguous, matches B). Two b128 loads.
static __device__ __forceinline__ v16bf load_a_frag(const unsigned short* A, int lda,
                                                    int m0, int kb, int lane) {
  int row = m0 + (lane & 15);
  int hi  = (lane >> 4) * 16;
  const unsigned short* p = A + (size_t)row * lda + kb + hi;
  union { v16bf v; us8 h[2]; } f;
  f.h[0] = *(const us8*)p;
  f.h[1] = *(const us8*)(p + 8);
  return f.v;
}

// B fragment for C = A * W^T with W row-major [N,K]: lane holds output column
// n0+(lane&15); its 16 K-values are contiguous in W's row -> 2x b128 loads.
static __device__ __forceinline__ v16bf load_bt_frag(const unsigned short* W, int ldw,
                                                     int n0, int kb, int lane, int N) {
  int col = n0 + (lane & 15); if (col >= N) col = N - 1;
  int khi = (lane >> 4) * 16;
  const unsigned short* p = W + (size_t)col * ldw + kb + khi;
  union { v16bf v; us8 h[2]; } f;
  f.h[0] = *(const us8*)p;
  f.h[1] = *(const us8*)(p + 8);
  return f.v;
}

// B fragment for row-major B [K,N] (strided gather along K).
static __device__ __forceinline__ v16bf load_b_frag(const unsigned short* B, int ldb,
                                                    int n0, int kb, int lane, int N) {
  int col = n0 + (lane & 15); if (col >= N) col = N - 1;
  int kk  = kb + (lane >> 4) * 16;
  union { v16bf v; unsigned short u[16]; } f;
  const unsigned short* p = B + (size_t)kk * ldb + col;
  #pragma unroll
  for (int e = 0; e < 16; ++e) f.u[e] = p[(size_t)e * ldb];
  return f.v;
}

// ---------------------------------------------------------------------------
// Elementwise / pack kernels
// ---------------------------------------------------------------------------
__global__ void k_cast(const float* __restrict__ in, unsigned short* __restrict__ out,
                       long n) {
  long i = (long)blockIdx.x * 256 + threadIdx.x;
  if (i < n) out[i] = f2bf(in[i]);
}

__global__ void k_castT(const float* __restrict__ in, unsigned short* __restrict__ out,
                        int R, int C) {
  int i = blockIdx.x * 256 + threadIdx.x;
  if (i >= R * C) return;
  int r = i / C, c = i % C;
  out[(size_t)c * R + r] = f2bf(in[i]);
}

__global__ void k_copyf(const float* __restrict__ in, float* __restrict__ out, long n) {
  long i = (long)blockIdx.x * 256 + threadIdx.x;
  if (i < n) out[i] = in[i];
}

// conv weight [O=256][I=256][3] -> [tap][O][I] bf16
__global__ void k_conv_repack(const float* __restrict__ w, unsigned short* __restrict__ out) {
  int i = blockIdx.x * 256 + threadIdx.x;
  if (i >= 196608) return;
  int o = i / 768, rem = i % 768, ii = rem / 3, j = rem % 3;
  out[j * 65536 + o * 256 + ii] = f2bf(w[i]);
}

__global__ void k_add(const float* __restrict__ a, const float* __restrict__ b,
                      float* __restrict__ oF, unsigned short* __restrict__ oB, long n) {
  long i = (long)blockIdx.x * 256 + threadIdx.x;
  if (i < n) { float v = a[i] + b[i]; oF[i] = v; oB[i] = f2bf(v); }
}

// row LayerNorm over 256 cols; mode 0: z=a; 1: z=a*b+a+b; 2: z=relu(a)
__global__ void k_ln(const float* __restrict__ A, const float* __restrict__ B2,
                     const float* __restrict__ g, const float* __restrict__ bt,
                     float* __restrict__ outF, unsigned short* __restrict__ outB,
                     int mode) {
  __shared__ float p1[256], p2[256];
  size_t row = blockIdx.x; int tid = threadIdx.x;
  float a = A[row * 256 + tid];
  float z;
  if (mode == 1) { float c = B2[row * 256 + tid]; z = a * c + a + c; }
  else if (mode == 2) z = fmaxf(a, 0.f);
  else z = a;
  p1[tid] = z; p2[tid] = z * z; __syncthreads();
  for (int o = 128; o > 0; o >>= 1) {
    if (tid < o) { p1[tid] += p1[tid + o]; p2[tid] += p2[tid + o]; }
    __syncthreads();
  }
  float mean = p1[0] * (1.f / 256.f);
  float var  = p2[0] * (1.f / 256.f) - mean * mean;
  float r = (z - mean) * rsqrtf(var + 1e-5f) * g[tid] + bt[tid];
  if (outF) outF[row * 256 + tid] = r;
  if (outB) outB[row * 256 + tid] = f2bf(r);
}

// deg^-1/2 of (adj + I)
__global__ void k_deg(const float* __restrict__ adj, float* __restrict__ dis) {
  __shared__ float part[256];
  int n = blockIdx.x, tid = threadIdx.x;
  float s = 0.f;
  for (int m = tid; m < 1024; m += 256) s += adj[(size_t)n * 1024 + m];
  if (tid == 0) s += 1.f;                     // +I diagonal
  part[tid] = s; __syncthreads();
  for (int o = 128; o > 0; o >>= 1) { if (tid < o) part[tid] += part[tid + o]; __syncthreads(); }
  if (tid == 0) { float d = part[0]; dis[n] = d > 0.f ? rsqrtf(d) : 0.f; }
}

__global__ void k_an(const float* __restrict__ adj, const float* __restrict__ dis,
                     unsigned short* __restrict__ an) {
  long i = (long)blockIdx.x * 256 + threadIdx.x;
  if (i >= 1048576) return;
  int n = (int)(i >> 10), m = (int)(i & 1023);
  float v = adj[i] + (n == m ? 1.f : 0.f);
  an[i] = f2bf(dis[n] * v * dis[m]);
}

// ---------------------------------------------------------------------------
// WMMA GEMM, C = A @ W^T (+bias, act). A bf16 [M,K] row-major, W bf16 [N,K].
// Block = 8 waves; each wave owns a 32x32 tile (2x2 of 16x16 accumulators).
// Tile coords pass through readfirstlane -> scalar branches, EXEC stays all-1.
// ---------------------------------------------------------------------------
__global__ void __launch_bounds__(256)
k_gemm_abt(const unsigned short* __restrict__ A, int lda, long long sA,
           const unsigned short* __restrict__ W, int ldw,
           const float* __restrict__ bias,
           float* __restrict__ C, unsigned short* __restrict__ Cb,
           int ldc, long long sC,
           int M, int N, int K, int flags) {
  int tid = threadIdx.x, wid = tid >> 5, lane = tid & 31;
  int lane15 = lane & 15, hi = lane >> 4;
  int wm = wid & 3, wn = wid >> 2;
  int m0 = rfl(blockIdx.y * 128 + wm * 32);
  int n0 = rfl(blockIdx.x * 64 + wn * 32);
  if (m0 >= M || n0 >= N) return;           // scalar branch, EXEC untouched
  const unsigned short* Ab = A + (size_t)blockIdx.z * sA;
  int mT = rfl((m0 + 16 < M) ? 2 : 1);
  int nT = rfl((n0 + 16 < N) ? 2 : 1);
  v8f acc[2][2];
  #pragma unroll
  for (int i = 0; i < 2; ++i)
    #pragma unroll
    for (int j = 0; j < 2; ++j)
      #pragma unroll
      for (int r = 0; r < 8; ++r) acc[i][j][r] = 0.f;
  for (int kb = 0; kb < K; kb += 32) {
    if (kb + 64 < K) {                       // pull next-next K slab into L1/L2
      __builtin_prefetch(Ab + (size_t)(m0 + lane15) * lda + kb + 64, 0, 1);
      __builtin_prefetch(W + (size_t)(n0 + lane15) * ldw + kb + 64, 0, 1);
    }
    v16bf a0 = load_a_frag(Ab, lda, m0, kb, lane);
    v16bf a1 = a0; if (mT > 1) a1 = load_a_frag(Ab, lda, m0 + 16, kb, lane);
    v16bf b0 = load_bt_frag(W, ldw, n0, kb, lane, N);
    v16bf b1 = b0; if (nT > 1) b1 = load_bt_frag(W, ldw, n0 + 16, kb, lane, N);
    acc[0][0] = wmma_bf(a0, b0, acc[0][0]);
    if (nT > 1) acc[0][1] = wmma_bf(a0, b1, acc[0][1]);
    if (mT > 1) {
      acc[1][0] = wmma_bf(a1, b0, acc[1][0]);
      if (nT > 1) acc[1][1] = wmma_bf(a1, b1, acc[1][1]);
    }
  }
  for (int i = 0; i < mT; ++i)
    for (int j = 0; j < nT; ++j) {
      int col = n0 + j * 16 + lane15;
      bool cok = col < N;
      float bv = (cok && (flags & GF_BIAS)) ? bias[col] : 0.f;
      #pragma unroll
      for (int r = 0; r < 8; ++r) {
        int row = m0 + i * 16 + r + hi * 8;
        if (!cok || row >= M) continue;
        float v = acc[i][j][r] + bv;
        if (flags & GF_RELU)     v = fmaxf(v, 0.f);
        if (flags & GF_SIGMOID)  v = sigm(v);
        if (flags & GF_SOFTPLUS) v = (v > 20.f) ? v : log1pf(expf(v));
        size_t off = (size_t)blockIdx.z * sC + (size_t)row * ldc + col;
        if (flags & GF_OUTF32)  C[off]  = v;
        if (flags & GF_OUTBF16) Cb[off] = f2bf(v);
      }
    }
}

// C = A @ B (+bias), B bf16 row-major [K,N]; batched over grid.z.
__global__ void __launch_bounds__(256)
k_gemm_ab(const unsigned short* __restrict__ A, int lda, long long sA,
          const unsigned short* __restrict__ B, int ldb, long long sB,
          const float* __restrict__ bias,
          float* __restrict__ C, unsigned short* __restrict__ Cb,
          int ldc, long long sC,
          int M, int N, int K, int flags) {
  int tid = threadIdx.x, wid = tid >> 5, lane = tid & 31;
  int lane15 = lane & 15, hi = lane >> 4;
  int wm = wid & 3, wn = wid >> 2;
  int m0 = rfl(blockIdx.y * 128 + wm * 32);
  int n0 = rfl(blockIdx.x * 64 + wn * 32);
  if (m0 >= M || n0 >= N) return;
  const unsigned short* Ab = A + (size_t)blockIdx.z * sA;
  const unsigned short* Bb = B + (size_t)blockIdx.z * sB;
  int mT = rfl((m0 + 16 < M) ? 2 : 1);
  int nT = rfl((n0 + 16 < N) ? 2 : 1);
  v8f acc[2][2];
  #pragma unroll
  for (int i = 0; i < 2; ++i)
    #pragma unroll
    for (int j = 0; j < 2; ++j)
      #pragma unroll
      for (int r = 0; r < 8; ++r) acc[i][j][r] = 0.f;
  for (int kb = 0; kb < K; kb += 32) {
    v16bf a0 = load_a_frag(Ab, lda, m0, kb, lane);
    v16bf a1 = a0; if (mT > 1) a1 = load_a_frag(Ab, lda, m0 + 16, kb, lane);
    v16bf b0 = load_b_frag(Bb, ldb, n0, kb, lane, N);
    v16bf b1 = b0; if (nT > 1) b1 = load_b_frag(Bb, ldb, n0 + 16, kb, lane, N);
    acc[0][0] = wmma_bf(a0, b0, acc[0][0]);
    if (nT > 1) acc[0][1] = wmma_bf(a0, b1, acc[0][1]);
    if (mT > 1) {
      acc[1][0] = wmma_bf(a1, b0, acc[1][0]);
      if (nT > 1) acc[1][1] = wmma_bf(a1, b1, acc[1][1]);
    }
  }
  for (int i = 0; i < mT; ++i)
    for (int j = 0; j < nT; ++j) {
      int col = n0 + j * 16 + lane15;
      bool cok = col < N;
      float bv = (cok && (flags & GF_BIAS)) ? bias[col] : 0.f;
      #pragma unroll
      for (int r = 0; r < 8; ++r) {
        int row = m0 + i * 16 + r + hi * 8;
        if (!cok || row >= M) continue;
        float v = acc[i][j][r] + bv;
        if (flags & GF_RELU) v = fmaxf(v, 0.f);
        size_t off = (size_t)blockIdx.z * sC + (size_t)row * ldc + col;
        if (flags & GF_OUTF32)  C[off]  = v;
        if (flags & GF_OUTBF16) Cb[off] = f2bf(v);
      }
    }
}

// ---------------------------------------------------------------------------
// Flash attention: 4 heads, hd=64, S=1024. One block per (b, head, 64-row
// q-tile). Q and K tiles are staged by the Tensor Data Mover (2D tile DMA,
// global->LDS, overlapped with the manual V-transpose staging) when the TDM
// builtin exists; S=QK^T and O+=PV via WMMA with online softmax.
// LDS carve: Qs@0, Ks@8192, VsT@16384 (byte offsets used by the descriptors).
// ---------------------------------------------------------------------------
__global__ void __launch_bounds__(256)
k_attn(const unsigned short* __restrict__ qkv,   // [B*1024, 768] bf16
       unsigned short* __restrict__ AOb) {       // [B*1024, 256] bf16
  extern __shared__ char smem[];
  unsigned short* Qs  = (unsigned short*)smem;       // [64][64]   @ 0
  unsigned short* Ks  = Qs + 4096;                   // [64][64]   @ 8192
  unsigned short* VsT = Ks + 4096;                   // [64][64]   @ 16384
  float*          Sb  = (float*)(VsT + 4096);        // [64][64]
  unsigned short* Ps  = (unsigned short*)(Sb + 4096);// [64][64]
  float* rowm = (float*)(Ps + 4096);
  float* rowl = rowm + 64;
  float* ralp = rowl + 64;

  const int tid = threadIdx.x, lane = tid & 31;
  const int lane15 = lane & 15, hi = lane >> 4;
  const int wid = tid >> 5;
  const int wm = rfl(wid & 3), wn = rfl(wid >> 2);
  const int q0 = blockIdx.x * 64, h = blockIdx.y, b = blockIdx.z;

#ifdef HAVE_TDM
  if (wid == 0)                                      // one DMA for the Q tile
    tdm_load_2d(0u, qkv + ((size_t)(b * 1024 + q0)) * 768 + h * 64,
                64, 64, 64, 1024, 768ull);
#else
  for (int c = tid; c < 512; c += 256) {             // stage Q (64x64, b128)
    int row = c >> 3, cc = c & 7;
    *(us8*)(Qs + row * 64 + cc * 8) =
        *(const us8*)(qkv + ((size_t)(b * 1024 + q0 + row)) * 768 + h * 64 + cc * 8);
  }
#endif
  if (tid < 64) { rowm[tid] = -1e30f; rowl[tid] = 0.f; }

  v8f oa[2];
  #pragma unroll
  for (int j = 0; j < 2; ++j)
    #pragma unroll
    for (int r = 0; r < 8; ++r) oa[j][r] = 0.f;

  for (int kt = 0; kt < 16; ++kt) {
    int k0 = kt * 64;
#ifdef HAVE_TDM
    if (wid == 0)                                    // K tile via TDM
      tdm_load_2d(8192u, qkv + 256 + ((size_t)(b * 1024 + k0)) * 768 + h * 64,
                  64, 64, 64, 1024, 768ull);
    for (int c = tid; c < 512; c += 256) {           // V^T staged manually
      int row = c >> 3, cc = c & 7;
      us8 vv = *(const us8*)(qkv + 512 +
                             ((size_t)(b * 1024 + k0 + row)) * 768 + h * 64 + cc * 8);
      #pragma unroll
      for (int e = 0; e < 8; ++e) VsT[(cc * 8 + e) * 64 + row] = vv[e];
    }
    if (wid == 0) tdm_wait();                        // s_wait_tensorcnt 0
#else
    for (int c = tid; c < 512; c += 256) {           // stage K and V^T
      int row = c >> 3, cc = c & 7;
      size_t base = ((size_t)(b * 1024 + k0 + row)) * 768 + h * 64 + cc * 8;
      *(us8*)(Ks + row * 64 + cc * 8) = *(const us8*)(qkv + 256 + base);
      us8 vv = *(const us8*)(qkv + 512 + base);
      #pragma unroll
      for (int e = 0; e < 8; ++e) VsT[(cc * 8 + e) * 64 + row] = vv[e];
    }
#endif
    __syncthreads();

    // S tile = Q @ K^T * 1/sqrt(64)
    v8f sa[2];
    #pragma unroll
    for (int j = 0; j < 2; ++j)
      #pragma unroll
      for (int r = 0; r < 8; ++r) sa[j][r] = 0.f;
    for (int ks = 0; ks < 64; ks += 32) {
      v16bf aq = load_a_frag(Qs, 64, wm * 16, ks, lane);
      #pragma unroll
      for (int j = 0; j < 2; ++j) {
        v16bf bk = load_bt_frag(Ks, 64, (wn * 2 + j) * 16, ks, lane, 64);
        sa[j] = wmma_bf(aq, bk, sa[j]);
      }
    }
    #pragma unroll
    for (int j = 0; j < 2; ++j) {
      int col = (wn * 2 + j) * 16 + lane15;
      #pragma unroll
      for (int r = 0; r < 8; ++r)
        Sb[(wm * 16 + r + hi * 8) * 64 + col] = sa[j][r] * 0.125f;
    }
    __syncthreads();

    if (tid < 64) {                                  // online softmax per row
      int row = tid;
      float mold = rowm[row], mx = mold;
      for (int c = 0; c < 64; ++c) mx = fmaxf(mx, Sb[row * 64 + c]);
      float al = expf(mold - mx), sum = 0.f;
      for (int c = 0; c < 64; ++c) {
        float p = expf(Sb[row * 64 + c] - mx);
        Ps[row * 64 + c] = f2bf(p);
        sum += p;
      }
      rowm[row] = mx;
      rowl[row] = rowl[row] * al + sum;
      ralp[row] = al;
    }
    __syncthreads();

    #pragma unroll
    for (int j = 0; j < 2; ++j)                      // rescale running O
      #pragma unroll
      for (int r = 0; r < 8; ++r) oa[j][r] *= ralp[wm * 16 + r + hi * 8];

    for (int ks = 0; ks < 64; ks += 32) {            // O += P @ V
      v16bf ap = load_a_frag(Ps, 64, wm * 16, ks, lane);
      #pragma unroll
      for (int j = 0; j < 2; ++j) {
        v16bf bv = load_bt_frag(VsT, 64, (wn * 2 + j) * 16, ks, lane, 64);
        oa[j] = wmma_bf(ap, bv, oa[j]);
      }
    }
    __syncthreads();
  }

  #pragma unroll
  for (int j = 0; j < 2; ++j) {
    int d0 = (wn * 2 + j) * 16 + lane15;
    #pragma unroll
    for (int r = 0; r < 8; ++r) {
      int rowin = wm * 16 + r + hi * 8;
      float v = oa[j][r] / fmaxf(rowl[rowin], 1e-20f);
      AOb[((size_t)(b * 1024 + q0 + rowin)) * 256 + h * 64 + d0] = f2bf(v);
    }
  }
}

// ---------------------------------------------------------------------------
// Persistent LayerNorm-LSTM: 2 layers x 256 steps, one 32-wave workgroup.
// Gate GEMM (16x1024x512) via WMMA each step; h/c/gates LDS-resident.
// ---------------------------------------------------------------------------
__global__ void __launch_bounds__(1024)
k_lstm(const unsigned short* __restrict__ xT,    // [B*T, 256] bf16 temporal emb
       const unsigned short* __restrict__ Wg,    // [2][1024][512] packed gates
       const float* __restrict__ bg,             // [2][1024]
       const float* __restrict__ lncg0, const float* __restrict__ lncb0,
       const float* __restrict__ lnhg0, const float* __restrict__ lnhb0,
       const float* __restrict__ lncg1, const float* __restrict__ lncb1,
       const float* __restrict__ lnhg1, const float* __restrict__ lnhb1,
       unsigned short* __restrict__ combG) {     // [B*T, 1024], cols 0..255
  extern __shared__ char smem[];
  unsigned short* comb = (unsigned short*)smem;  // [16][512]
  float* gates = (float*)(comb + 8192);          // [16][1024]
  float* hS    = gates + 16384;                  // [2][16][256]
  float* cS    = hS + 8192;                      // [2][16][256]
  float* craw  = cS + 8192;                      // [16][256]
  float* hraw  = craw + 4096;                    // [16][256]
  const int tid = threadIdx.x, wid = tid >> 5, lane = tid & 31;
  const int lane15 = lane & 15, hi = lane >> 4;

  for (int i = tid; i < 8192; i += 1024) { hS[i] = 0.f; cS[i] = 0.f; }
  __syncthreads();

  for (int t = 0; t < 256; ++t) {
    for (int l = 0; l < 2; ++l) {
      for (int i = tid; i < 8192; i += 1024) {   // comb = [x_t | h_prev] bf16
        int rb = i >> 9, cc = i & 511;
        unsigned short v;
        if (cc < 256)
          v = (l == 0) ? xT[((size_t)(rb * 256 + t)) * 256 + cc]
                       : f2bf(hS[rb * 256 + cc]);           // layer0 h_t
        else
          v = f2bf(hS[l * 4096 + rb * 256 + (cc - 256)]);   // own h_{t-1}
        comb[rb * 512 + cc] = v;
      }
      __syncthreads();

      const unsigned short* Wl = Wg + (size_t)l * 1024 * 512;
      for (int j = 0; j < 2; ++j) {              // 32 waves x 2 n-tiles = 64
        int n0 = rfl((wid * 2 + j) * 16);
        v8f acc;
        #pragma unroll
        for (int r = 0; r < 8; ++r) acc[r] = 0.f;
        for (int kb = 0; kb < 512; kb += 32) {
          v16bf a = load_a_frag(comb, 512, 0, kb, lane);
          v16bf w = load_bt_frag(Wl, 512, n0, kb, lane, 1024);
          acc = wmma_bf(a, w, acc);
        }
        #pragma unroll
        for (int r = 0; r < 8; ++r) {
          int row = r + hi * 8, col = n0 + lane15;
          gates[row * 1024 + col] = acc[r] + bg[l * 1024 + col];
        }
      }
      __syncthreads();

      for (int i = tid; i < 4096; i += 1024) {   // c_raw
        int rb = i >> 8, hc = i & 255;
        float iv = sigm(gates[rb * 1024 + hc]);
        float fv = sigm(gates[rb * 1024 + 256 + hc]);
        float gv = tanhf(gates[rb * 1024 + 768 + hc]);
        craw[i] = fv * cS[l * 4096 + i] + iv * gv;
      }
      __syncthreads();

      if (wid < 16) {                            // wave-per-row LayerNorms
        int rb = wid;
        const float* lcg = l ? lncg1 : lncg0; const float* lcb = l ? lncb1 : lncb0;
        const float* lhg = l ? lnhg1 : lnhg0; const float* lhb = l ? lnhb1 : lnhb0;
        float s = 0.f, ss = 0.f;
        #pragma unroll
        for (int e = 0; e < 8; ++e) { float v = craw[rb*256 + lane*8 + e]; s += v; ss += v*v; }
        for (int o = 16; o > 0; o >>= 1) { s += __shfl_down(s, o, 32); ss += __shfl_down(ss, o, 32); }
        s = __shfl(s, 0, 32); ss = __shfl(ss, 0, 32);
        float mean = s * (1.f/256.f), var = ss * (1.f/256.f) - mean * mean;
        float inv = rsqrtf(var + 1e-5f);
        #pragma unroll
        for (int e = 0; e < 8; ++e) {
          int col = lane * 8 + e, i = rb * 256 + col;
          float cn = (craw[i] - mean) * inv * lcg[col] + lcb[col];
          cS[l * 4096 + i] = cn;
          hraw[i] = sigm(gates[rb * 1024 + 512 + col]) * tanhf(cn);
        }
        s = 0.f; ss = 0.f;
        #pragma unroll
        for (int e = 0; e < 8; ++e) { float v = hraw[rb*256 + lane*8 + e]; s += v; ss += v*v; }
        for (int o = 16; o > 0; o >>= 1) { s += __shfl_down(s, o, 32); ss += __shfl_down(ss, o, 32); }
        s = __shfl(s, 0, 32); ss = __shfl(ss, 0, 32);
        mean = s * (1.f/256.f); var = ss * (1.f/256.f) - mean * mean;
        inv = rsqrtf(var + 1e-5f);
        #pragma unroll
        for (int e = 0; e < 8; ++e) {
          int col = lane * 8 + e, i = rb * 256 + col;
          hS[l * 4096 + i] = (hraw[i] - mean) * inv * lhg[col] + lhb[col];
        }
      }
      __syncthreads();
    }
    for (int i = tid; i < 4096; i += 1024) {     // emit layer-1 h_t
      int rb = i >> 8, hc = i & 255;
      combG[((size_t)(rb * 256 + t)) * 1024 + hc] = f2bf(hS[4096 + i]);
    }
    __syncthreads();
  }
}

// ---------------------------------------------------------------------------
// Dilated conv1d (kernel 3) as 3 time-shifted WMMA GEMMs; relu(.+b) written
// straight into the concat buffer column block.
// ---------------------------------------------------------------------------
__global__ void __launch_bounds__(256)
k_dconv(const unsigned short* __restrict__ X,    // combG cols 0..255, ld 1024
        const unsigned short* __restrict__ Wt,   // [3][256][256] bf16
        const float* __restrict__ bias,
        unsigned short* __restrict__ outB,       // combG + colOff, ld 1024
        int dil) {
  int tid = threadIdx.x, wid = tid >> 5, lane = tid & 31;
  int lane15 = lane & 15, hi = lane >> 4;
  int wm = wid & 3, wn = wid >> 2;
  int m0 = rfl(blockIdx.y * 128 + wm * 32);
  int n0 = rfl(blockIdx.x * 64 + wn * 32);
  v8f acc[2][2];
  #pragma unroll
  for (int i = 0; i < 2; ++i)
    #pragma unroll
    for (int j = 0; j < 2; ++j)
      #pragma unroll
      for (int r = 0; r < 8; ++r) acc[i][j][r] = 0.f;
  for (int tap = 0; tap < 3; ++tap) {
    int shift = dil * (tap - 1);
    const unsigned short* Wj = Wt + tap * 65536;
    for (int kb = 0; kb < 256; kb += 32) {
      v16bf a[2];
      #pragma unroll
      for (int i = 0; i < 2; ++i) {
        int rl = m0 + i * 16 + lane15;
        int bb = rl >> 8, tt = rl & 255, t2 = tt + shift;
        bool ok = (t2 >= 0) && (t2 < 256);
        const unsigned short* p =
            X + ((size_t)(bb * 256 + (ok ? t2 : 0))) * 1024 + kb + hi * 16;
        union { v16bf v; us8 h[2]; } f;
        us8 z = {0, 0, 0, 0, 0, 0, 0, 0};
        f.h[0] = ok ? *(const us8*)p : z;
        f.h[1] = ok ? *(const us8*)(p + 8) : z;
        a[i] = f.v;
      }
      v16bf b0 = load_bt_frag(Wj, 256, n0, kb, lane, 256);
      v16bf b1 = load_bt_frag(Wj, 256, n0 + 16, kb, lane, 256);
      acc[0][0] = wmma_bf(a[0], b0, acc[0][0]);
      acc[0][1] = wmma_bf(a[0], b1, acc[0][1]);
      acc[1][0] = wmma_bf(a[1], b0, acc[1][0]);
      acc[1][1] = wmma_bf(a[1], b1, acc[1][1]);
    }
  }
  #pragma unroll
  for (int i = 0; i < 2; ++i)
    #pragma unroll
    for (int j = 0; j < 2; ++j) {
      int col = n0 + j * 16 + lane15;
      float bv = bias[col];
      #pragma unroll
      for (int r = 0; r < 8; ++r) {
        int row = m0 + i * 16 + r + hi * 8;
        float v = fmaxf(acc[i][j][r] + bv, 0.f);
        outB[(size_t)row * 1024 + col] = f2bf(v);
      }
    }
}

// ---------------------------------------------------------------------------
// Softmax attention pooling: agg[b,:] = sum_s softmax_s(x.w + b) * x[b,s,:]
// ---------------------------------------------------------------------------
__global__ void k_pool(const float* __restrict__ X, int S,
                       const float* __restrict__ wv, const float* __restrict__ bsc,
                       float* __restrict__ combF, unsigned short* __restrict__ combB,
                       int colOff) {
  __shared__ float lg[1024];
  __shared__ float part[256];
  int b = blockIdx.x, tid = threadIdx.x;
  for (int s = tid; s < S; s += 256) {
    const float* xr = X + ((size_t)b * S + s) * 256;
    float acc = 0.f;
    for (int d = 0; d < 256; ++d) acc += xr[d] * wv[d];
    lg[s] = acc + bsc[0];
  }
  __syncthreads();
  float m = -1e30f;
  for (int s = tid; s < S; s += 256) m = fmaxf(m, lg[s]);
  part[tid] = m; __syncthreads();
  for (int o = 128; o > 0; o >>= 1) { if (tid < o) part[tid] = fmaxf(part[tid], part[tid + o]); __syncthreads(); }
  float mx = part[0]; __syncthreads();
  float sm = 0.f;
  for (int s = tid; s < S; s += 256) sm += expf(lg[s] - mx);
  part[tid] = sm; __syncthreads();
  for (int o = 128; o > 0; o >>= 1) { if (tid < o) part[tid] += part[tid + o]; __syncthreads(); }
  float inv = 1.f / part[0]; __syncthreads();
  for (int s = tid; s < S; s += 256) lg[s] = expf(lg[s] - mx) * inv;
  __syncthreads();
  float acc = 0.f;
  for (int s = 0; s < S; ++s) acc += lg[s] * X[((size_t)b * S + s) * 256 + tid];
  combF[b * 512 + colOff + tid] = acc;
  combB[b * 512 + colOff + tid] = f2bf(acc);
}

__global__ void k_softmax_case(const float* __restrict__ lg, float* __restrict__ out) {
  int r = threadIdx.x;
  if (r >= 16) return;
  float mx = -1e30f;
  for (int c = 0; c < 16; ++c) mx = fmaxf(mx, lg[r * 16 + c]);
  float s = 0.f;
  for (int c = 0; c < 16; ++c) s += expf(lg[r * 16 + c] - mx);
  for (int c = 0; c < 16; ++c) out[r * 16 + c] = expf(lg[r * 16 + c] - mx) / s;
}

// ---------------------------------------------------------------------------
// Host orchestration
// ---------------------------------------------------------------------------
enum {
  IN_XS = 0, IN_ADJ, IN_XT,
  E_LOCW, E_LOCB, E_CTXW, E_CTXB, E_LNG, E_LNB,             // 3..8
  G_W, G_B, G_LNG, G_LNB, G_INW, G_INB, G_OUTW, G_OUTB,     // 9..16
  L0 = 17,   // wi,bi,wf,bf,wo,bo,wg,bg,lncg,lncb,lnhg,lnhb
  L1 = 29,
  CV0W = 41, CV0B, CV1W, CV1B, CV2W, CV2B,
  P_PROJW = 47, P_PROJB,
  P_SAW = 49, P_SAB, P_TAW, P_TAB,
  P_C1W = 53, P_C1B, P_C2W, P_C2B,
  P_R1W = 57, P_R1B, P_R2W, P_R2B,
  P_K1W = 61, P_K1B, P_K2W, P_K2B
};

extern "C" void kernel_launch(void* const* d_in, const int* in_sizes, int n_in,
                              void* d_out, int out_size, void* d_ws, size_t ws_size,
                              hipStream_t stream) {
  (void)in_sizes; (void)n_in; (void)out_size; (void)ws_size;
  const float* F[66];
  for (int i = 0; i < 65; ++i) F[i] = (const float*)d_in[i];
  float* out = (float*)d_out;   // [case 16x16][r0 16][risk 16x1024]

  char* ws = (char*)d_ws;
  size_t off = 0;
  auto alloc = [&](size_t bytes) -> char* {
    off = (off + 255) & ~(size_t)255;
    char* p = ws + off; off += bytes; return p;
  };
  auto aU = [&](size_t n) { return (unsigned short*)alloc(n * 2); };
  auto aF = [&](size_t n) { return (float*)alloc(n * 4); };

  // weights (bf16) ---------------------------------------------------------
  unsigned short* W_LOC  = aU(65536);
  unsigned short* W_CTX  = aU(65536);
  unsigned short* W_INW  = aU(196608);
  unsigned short* W_OUTW = aU(65536);
  unsigned short* W_GCNT = aU(65536);        // gcn.W transposed -> [H,D]
  unsigned short* W_GATES= aU(1048576);      // [2][1024][512]
  float*          B_GATES= aF(2048);         // [2][1024]
  unsigned short* W_CONV = aU(589824);       // 3 x [3][256][256]
  unsigned short* W_PROJ = aU(262144);
  unsigned short* W_C1   = aU(131072);
  unsigned short* W_C2   = aU(4096);
  unsigned short* W_R1   = aU(65536);
  unsigned short* W_R2   = aU(128);
  unsigned short* W_K1   = aU(131072);
  unsigned short* W_K2   = aU(262144);
  // activations ------------------------------------------------------------
  unsigned short* XS_B   = aU(4194304);      // [16384,256]
  float*          LOC_F  = aF(4194304);
  float*          CTX_F  = aF(4194304);
  float*          EMBS_F = aF(4194304);
  unsigned short* EMBS_B = aU(4194304);
  unsigned short* QKV_B  = aU(12582912);     // [16384,768]
  unsigned short* AO_B   = aU(4194304);
  float*          PRJ_F  = aF(4194304);
  float*          XC_F   = aF(4194304);
  unsigned short* XC_B   = aU(4194304);
  unsigned short* XW_B   = aU(4194304);
  float*          GOUT_F = aF(4194304);
  float*          SP_F   = aF(4194304);
  unsigned short* AN_B   = aU(1048576);
  float*          DIS_F  = aF(1024);
  unsigned short* XT_B   = aU(1048576);      // [4096,256]
  unsigned short* TEMB_B = aU(1048576);
  float*          TLOC_F = aF(1048576);
  float*          TCTX_F = aF(1048576);
  unsigned short* COMBG  = aU(4194304);      // [4096,1024] concat
  float*          TP_F   = aF(1048576);
  float*          COMBF  = aF(8192);         // [16,512]
  unsigned short* COMBB  = aU(8192);
  unsigned short* H1_B   = aU(4096);
  unsigned short* HR_B   = aU(2048);
  unsigned short* HK_B   = aU(4096);
  float*          LOG_F  = aF(256);

  auto cast = [&](int idx, unsigned short* dst, long n) {
    k_cast<<<dim3((unsigned)((n + 255) / 256)), dim3(256), 0, stream>>>(F[idx], dst, n);
  };
  auto gemm_abt = [&](const unsigned short* A, int lda, const unsigned short* W,
                      int ldw, const float* bias, float* C, unsigned short* Cb,
                      int ldc, int M, int N, int K, int flags) {
    dim3 g((N + 63) / 64, (M + 127) / 128, 1);
    k_gemm_abt<<<g, 256, 0, stream>>>(A, lda, 0ll, W, ldw, bias, C, Cb, ldc, 0ll,
                                      M, N, K, flags);
  };

  // ---- weight packing (per call; deterministic) --------------------------
  cast(IN_XS, XS_B, 4194304);
  cast(IN_XT, XT_B, 1048576);
  cast(E_LOCW, W_LOC, 65536);
  cast(E_CTXW, W_CTX, 65536);
  cast(G_INW, W_INW, 196608);
  cast(G_OUTW, W_OUTW, 65536);
  k_castT<<<dim3(256), dim3(256), 0, stream>>>(F[G_W], W_GCNT, 256, 256);
  for (int l = 0; l < 2; ++l) {
    int base = l ? L1 : L0;
    for (int gidx = 0; gidx < 4; ++gidx) {     // i,f,o,g
      cast(base + 2 * gidx, W_GATES + (size_t)l * 524288 + (size_t)gidx * 131072, 131072);
      k_copyf<<<dim3(1), dim3(256), 0, stream>>>(F[base + 2 * gidx + 1],
                                                 B_GATES + l * 1024 + gidx * 256, 256);
    }
  }
  for (int ci = 0; ci < 3; ++ci)
    k_conv_repack<<<dim3(768), dim3(256), 0, stream>>>(F[CV0W + 2 * ci],
                                                       W_CONV + (size_t)ci * 196608);
  cast(P_PROJW, W_PROJ, 262144);
  cast(P_C1W, W_C1, 131072);
  cast(P_C2W, W_C2, 4096);
  cast(P_R1W, W_R1, 65536);
  cast(P_R2W, W_R2, 128);
  cast(P_K1W, W_K1, 131072);
  cast(P_K2W, W_K2, 262144);

  // ---- spatial branch ----------------------------------------------------
  gemm_abt(XS_B, 256, W_LOC, 256, F[E_LOCB], LOC_F, nullptr, 256,
           16384, 256, 256, GF_BIAS | GF_OUTF32);
  gemm_abt(XS_B, 256, W_CTX, 256, F[E_CTXB], CTX_F, nullptr, 256,
           16384, 256, 256, GF_BIAS | GF_OUTF32);
  k_ln<<<dim3(16384), dim3(256), 0, stream>>>(LOC_F, CTX_F, F[E_LNG], F[E_LNB],
                                              EMBS_F, EMBS_B, 1);
  gemm_abt(EMBS_B, 256, W_INW, 256, F[G_INB], nullptr, QKV_B, 768,
           16384, 768, 256, GF_BIAS | GF_OUTBF16);
  {
    size_t smem = 3 * 4096 * 2 + 4096 * 4 + 4096 * 2 + 3 * 64 * 4;  // 49920 B
    k_attn<<<dim3(16, 4, 16), dim3(256), smem, stream>>>(QKV_B, AO_B);
  }
  gemm_abt(AO_B, 256, W_OUTW, 256, F[G_OUTB], PRJ_F, nullptr, 256,
           16384, 256, 256, GF_BIAS | GF_OUTF32);
  k_add<<<dim3(16384), dim3(256), 0, stream>>>(EMBS_F, PRJ_F, XC_F, XC_B, 4194304);
  gemm_abt(XC_B, 256, W_GCNT, 256, nullptr, nullptr, XW_B, 256,
           16384, 256, 256, GF_OUTBF16);
  k_deg<<<dim3(1024), dim3(256), 0, stream>>>(F[IN_ADJ], DIS_F);
  k_an<<<dim3(4096), dim3(256), 0, stream>>>(F[IN_ADJ], DIS_F, AN_B);
  {   // out[b] = AN @ (xc W)[b] + b  : batched over 16
    dim3 g(4, 8, 16);
    k_gemm_ab<<<g, 256, 0, stream>>>(AN_B, 1024, 0ll, XW_B, 256, 262144ll,
                                     F[G_B], GOUT_F, nullptr, 256, 262144ll,
                                     1024, 256, 1024, GF_BIAS | GF_OUTF32);
  }
  k_ln<<<dim3(16384), dim3(256), 0, stream>>>(GOUT_F, nullptr, F[G_LNG], F[G_LNB],
                                              SP_F, nullptr, 2);

  // ---- temporal branch ---------------------------------------------------
  gemm_abt(XT_B, 256, W_LOC, 256, F[E_LOCB], TLOC_F, nullptr, 256,
           4096, 256, 256, GF_BIAS | GF_OUTF32);
  gemm_abt(XT_B, 256, W_CTX, 256, F[E_CTXB], TCTX_F, nullptr, 256,
           4096, 256, 256, GF_BIAS | GF_OUTF32);
  k_ln<<<dim3(4096), dim3(256), 0, stream>>>(TLOC_F, TCTX_F, F[E_LNG], F[E_LNB],
                                             nullptr, TEMB_B, 1);
  k_lstm<<<dim3(1), dim3(1024), 180224, stream>>>(
      TEMB_B, W_GATES, B_GATES,
      F[L0 + 8], F[L0 + 9], F[L0 + 10], F[L0 + 11],
      F[L1 + 8], F[L1 + 9], F[L1 + 10], F[L1 + 11], COMBG);
  for (int ci = 0; ci < 3; ++ci)
    k_dconv<<<dim3(4, 32), dim3(256), 0, stream>>>(
        COMBG, W_CONV + (size_t)ci * 196608, F[CV0B + 2 * ci],
        COMBG + 256 * (1 + ci), 1 << ci);
  gemm_abt(COMBG, 1024, W_PROJ, 1024, F[P_PROJB], TP_F, nullptr, 256,
           4096, 256, 1024, GF_BIAS | GF_OUTF32);

  // ---- pooling + heads ---------------------------------------------------
  k_pool<<<dim3(16), dim3(256), 0, stream>>>(SP_F, 1024, F[P_SAW], F[P_SAB],
                                             COMBF, COMBB, 0);
  k_pool<<<dim3(16), dim3(256), 0, stream>>>(TP_F, 256, F[P_TAW], F[P_TAB],
                                             COMBF, COMBB, 256);
  gemm_abt(COMBB, 512, W_C1, 512, F[P_C1B], nullptr, H1_B, 256,
           16, 256, 512, GF_BIAS | GF_RELU | GF_OUTBF16);
  gemm_abt(H1_B, 256, W_C2, 256, F[P_C2B], LOG_F, nullptr, 16,
           16, 16, 256, GF_BIAS | GF_OUTF32);
  k_softmax_case<<<dim3(1), dim3(64), 0, stream>>>(LOG_F, out);
  gemm_abt(COMBB, 512, W_R1, 512, F[P_R1B], nullptr, HR_B, 128,
           16, 128, 512, GF_BIAS | GF_RELU | GF_OUTBF16);
  gemm_abt(HR_B, 128, W_R2, 128, F[P_R2B], out + 256, nullptr, 1,
           16, 1, 128, GF_BIAS | GF_SOFTPLUS | GF_OUTF32);
  gemm_abt(COMBB, 512, W_K1, 512, F[P_K1B], nullptr, HK_B, 256,
           16, 256, 512, GF_BIAS | GF_RELU | GF_OUTBF16);
  gemm_abt(HK_B, 256, W_K2, 256, F[P_K2B], out + 272, nullptr, 1024,
           16, 1024, 256, GF_BIAS | GF_SIGMOID | GF_OUTF32);
}
